// MixtralMoE_13838384627728
// MI455X (gfx1250) — matrix operand order repaired and checked by reading, not compiled, not run on hardware
//
#include <hip/hip_runtime.h>
#include <hip/hip_bf16.h>
#include <cstdint>
#include <cstddef>

// ---------------------------------------------------------------------------
// Mixtral-style MoE for MI455X (gfx1250), wave32 + WMMA bf16.
// Compute-bound (0.82 TFLOP vs ~38us HBM floor) => deep-K
// v_wmma_f32_16x16x32_bf16, fp32 weights converted to bf16 while staging
// global->LDS (single fp32 HBM pass). Each wave register-blocks a 32x32
// output tile (2x2 subtiles) so one LDS fragment feeds two wmmas:
// 1.5 ds_load_b128 per wmma instead of 3.
// ---------------------------------------------------------------------------

typedef __bf16 v16bf __attribute__((ext_vector_type(16)));
typedef float  v8f   __attribute__((ext_vector_type(8)));

constexpr int BN = 4, TT = 2048, HH = 16, DHD = 128;
constexpr int N = BN * TT;      // 8192 tokens
constexpr int D = HH * DHD;     // 2048
constexpr int F = 4096;
constexpr int E = 8;
constexpr int TOPK = 2;

constexpr int TILE_M = 64;      // pairs per workgroup tile (M)
constexpr int TILE_N = 128;     // output cols per workgroup tile (N)
constexpr int TILE_K = 64;      // K chunk staged in LDS
constexpr int MAXPAIRS = N * TOPK + E * TILE_M;   // 16896 (segments padded to 64)
constexpr int MTILES = MAXPAIRS / TILE_M;         // 264

union FragBF  { uint4 q[2]; v16bf v; };
union Pack32  { __bf16 h[32]; uint4 q[4]; };

// --------------------------- 1. x -> bf16, zero counts ----------------------
__global__ __launch_bounds__(256) void prep_kernel(const float* __restrict__ x,
                                                   __bf16* __restrict__ xb,
                                                   int* __restrict__ cnt) {
    if (blockIdx.x == 0 && threadIdx.x < E) cnt[threadIdx.x] = 0;
    int idx = blockIdx.x * 256 + threadIdx.x;            // float4 index
    float4 v = ((const float4*)x)[idx];
    union { __bf16 h4[4]; uint2 u; } pk;
    pk.h4[0] = (__bf16)v.x; pk.h4[1] = (__bf16)v.y;
    pk.h4[2] = (__bf16)v.z; pk.h4[3] = (__bf16)v.w;
    *(uint2*)(xb + (size_t)idx * 4) = pk.u;
}

// --------------------------- 2. router (top-2 softmax) ----------------------
__global__ __launch_bounds__(256) void router_kernel(const float* __restrict__ x,
                                                     const float* __restrict__ gw,
                                                     int* __restrict__ topidx,
                                                     float* __restrict__ topw,
                                                     int* __restrict__ cnt) {
    const int wave = threadIdx.x >> 5, lane = threadIdx.x & 31;
    const int t = blockIdx.x * 8 + wave;
    const float* xr = x + (size_t)t * D;
    float logit[E];
    for (int e = 0; e < E; ++e) {
        const float* gr = gw + (size_t)e * D;
        float p = 0.f;
        for (int d = lane; d < D; d += 32) p += xr[d] * gr[d];
        for (int m = 16; m > 0; m >>= 1) p += __shfl_xor(p, m, 32);
        logit[e] = p;
    }
    if (lane == 0) {
        int i0 = 0; float l0 = logit[0];
        for (int e = 1; e < E; ++e) if (logit[e] > l0) { l0 = logit[e]; i0 = e; }
        int i1 = -1; float l1 = -3.4e38f;
        for (int e = 0; e < E; ++e) if (e != i0 && logit[e] > l1) { l1 = logit[e]; i1 = e; }
        float wg0 = 1.f / (1.f + __expf(l1 - l0));       // softmax over (l0,l1)
        topidx[2*t] = i0; topidx[2*t+1] = i1;
        topw[2*t] = wg0;  topw[2*t+1] = 1.f - wg0;
        atomicAdd(&cnt[i0], 1);
        atomicAdd(&cnt[i1], 1);
    }
}

// ------------------ 3. padded per-expert offsets + list init ----------------
__global__ __launch_bounds__(256) void offsets_kernel(const int* __restrict__ cnt,
                                                      int* __restrict__ offs,
                                                      int* __restrict__ cursor,
                                                      int* __restrict__ ptok,
                                                      float* __restrict__ pw) {
    if (threadIdx.x == 0) {
        int o = 0;
        for (int e = 0; e < E; ++e) {
            offs[e] = o; cursor[e] = o;
            o += (cnt[e] + TILE_M - 1) & ~(TILE_M - 1);  // pad: tiles never straddle experts
        }
        offs[E] = o;
    }
    __syncthreads();
    for (int i = threadIdx.x; i < MAXPAIRS; i += 256) { ptok[i] = -1; pw[i] = 0.f; }
}

// --------------------------- 4. scatter pair lists --------------------------
__global__ __launch_bounds__(256) void scatter_kernel(const int* __restrict__ topidx,
                                                      const float* __restrict__ topw,
                                                      int* __restrict__ cursor,
                                                      int* __restrict__ ptok,
                                                      float* __restrict__ pw,
                                                      int* __restrict__ ppos) {
    int t = blockIdx.x * 256 + threadIdx.x;
    for (int k = 0; k < TOPK; ++k) {
        int e = topidx[2*t + k];
        int pos = atomicAdd(&cursor[e], 1);
        ptok[pos] = t; pw[pos] = topw[2*t + k]; ppos[2*t + k] = pos;
    }
}

// ---- helper: stage 32 fp32 -> 32 bf16 into an LDS row chunk ----------------
__device__ __forceinline__ void stage_row32(const float* __restrict__ src,
                                            __bf16* __restrict__ dst) {
    Pack32 p;
    #pragma unroll
    for (int j = 0; j < 8; ++j) {
        float4 a = ((const float4*)src)[j];
        p.h[4*j+0] = (__bf16)a.x; p.h[4*j+1] = (__bf16)a.y;
        p.h[4*j+2] = (__bf16)a.z; p.h[4*j+3] = (__bf16)a.w;
    }
    #pragma unroll
    for (int j = 0; j < 4; ++j) ((uint4*)dst)[j] = p.q[j];
}

// --------- 5. grouped GEMM1: h = silu(x@w1^T) * (x@w3^T)  (bf16 WMMA) -------
__global__ __launch_bounds__(256) void gemm1_kernel(const __bf16* __restrict__ xb,
                                                    const float* __restrict__ w1,
                                                    const float* __restrict__ w3,
                                                    const int* __restrict__ offs,
                                                    const int* __restrict__ ptok,
                                                    __bf16* __restrict__ h) {
    __shared__ __bf16 lA [TILE_M][TILE_K];    //  8 KB
    __shared__ __bf16 lB1[TILE_N][TILE_K];    // 16 KB
    __shared__ __bf16 lB3[TILE_N][TILE_K];    // 16 KB
    __shared__ int stok[TILE_M];

    const int tid   = threadIdx.x;
    const int mBase = blockIdx.x * TILE_M;
    const int fBase = blockIdx.y * TILE_N;
    if (mBase >= offs[E]) return;
    int e = 0;
    while (e < E - 1 && mBase >= offs[e + 1]) ++e;

    if (tid < TILE_M) stok[tid] = ptok[mBase + tid];
    __syncthreads();

    const int wave = tid >> 5, lane = tid & 31;
    const int mSub = (wave >> 2) * 32, nSub = (wave & 3) * 32;   // 2x4 wave grid
    v8f acc1[2][2], acc3[2][2];
    #pragma unroll
    for (int mi = 0; mi < 2; ++mi)
        #pragma unroll
        for (int ni = 0; ni < 2; ++ni) {
            acc1[mi][ni] = (v8f){0,0,0,0,0,0,0,0};
            acc3[mi][ni] = (v8f){0,0,0,0,0,0,0,0};
        }

    const int ar = tid >> 2, ac = tid & 3;   // A staging: row 0..63, 16-elem chunk
    const int br = tid >> 1, bc = tid & 1;   // B staging: row 0..127, 32-elem chunk
    const size_t wrow = ((size_t)e * F + fBase + br) * D;

    for (int k0 = 0; k0 < D; k0 += TILE_K) {
        // stage A: gathered bf16 token rows (zeros for padding rows)
        {
            int tok = stok[ar];
            uint4 v0 = make_uint4(0u,0u,0u,0u), v1 = v0;
            if (tok >= 0) {
                const uint4* s = (const uint4*)(xb + (size_t)tok * D + k0 + ac * 16);
                v0 = s[0]; v1 = s[1];
            }
            uint4* d = (uint4*)&lA[ar][ac * 16];
            d[0] = v0; d[1] = v1;
        }
        // stage B1/B3: fp32 -> bf16 during global->LDS copy
        {
            const float* s1 = w1 + wrow + k0 + bc * 32;
            const float* s3 = w3 + wrow + k0 + bc * 32;
            if (k0 + TILE_K < D) {
                __builtin_prefetch(s1 + TILE_K, 0, 1);   // global_prefetch_b8
                __builtin_prefetch(s3 + TILE_K, 0, 1);
            }
            stage_row32(s1, &lB1[br][bc * 32]);
            stage_row32(s3, &lB3[br][bc * 32]);
        }
        __syncthreads();
        #pragma unroll
        for (int ks = 0; ks < TILE_K; ks += 32) {
            // A 16x32 bf16 layout: lanes 0-15 rows M=lane, K={0..7,16..23};
            //                      lanes 16-31 rows M=lane-16, K={8..15,24..31}
            const int akb = ks + ((lane & 16) ? 8 : 0);
            FragBF a[2];
            #pragma unroll
            for (int mi = 0; mi < 2; ++mi) {
                const int arow = mSub + mi * 16 + (lane & 15);
                a[mi].q[0] = *(const uint4*)&lA[arow][akb];
                a[mi].q[1] = *(const uint4*)&lA[arow][akb + 16];
            }
            // B 32x16 bf16 layout: lanes 0-15 N=lane K={0..7,16..23}; lanes 16-31 K={8..15,24..31}
            const int bkb = ks + ((lane & 16) ? 8 : 0);
            FragBF b1[2], b3[2];
            #pragma unroll
            for (int ni = 0; ni < 2; ++ni) {
                const int bcol = nSub + ni * 16 + (lane & 15);
                b1[ni].q[0] = *(const uint4*)&lB1[bcol][bkb];
                b1[ni].q[1] = *(const uint4*)&lB1[bcol][bkb + 16];
                b3[ni].q[0] = *(const uint4*)&lB3[bcol][bkb];
                b3[ni].q[1] = *(const uint4*)&lB3[bcol][bkb + 16];
            }
            #pragma unroll
            for (int mi = 0; mi < 2; ++mi)
                #pragma unroll
                for (int ni = 0; ni < 2; ++ni) {
                    acc1[mi][ni] = __builtin_amdgcn_wmma_f32_16x16x32_bf16(
                        false, a[mi].v, false, b1[ni].v, (short)0, acc1[mi][ni], false, false);
                    acc3[mi][ni] = __builtin_amdgcn_wmma_f32_16x16x32_bf16(
                        false, a[mi].v, false, b3[ni].v, (short)0, acc3[mi][ni], false, false);
                }
        }
        __syncthreads();
    }

    // epilogue: h = silu(gate) * up, stored bf16
    #pragma unroll
    for (int mi = 0; mi < 2; ++mi) {
        const int rowTop = mBase + mSub + mi * 16 + ((lane & 16) ? 8 : 0);
        #pragma unroll
        for (int ni = 0; ni < 2; ++ni) {
            const int col = fBase + nSub + ni * 16 + (lane & 15);
            #pragma unroll
            for (int r = 0; r < 8; ++r) {
                float g = acc1[mi][ni][r], u = acc3[mi][ni][r];
                float hv = (g / (1.f + __expf(-g))) * u;
                h[(size_t)(rowTop + r) * F + col] = (__bf16)hv;
            }
        }
    }
}

// ------------- 6. grouped GEMM2: y = we * (h @ w2^T)  (bf16 WMMA) -----------
__global__ __launch_bounds__(256) void gemm2_kernel(const __bf16* __restrict__ h,
                                                    const float* __restrict__ w2,
                                                    const int* __restrict__ offs,
                                                    const float* __restrict__ pw,
                                                    float* __restrict__ y) {
    __shared__ __bf16 lA[TILE_M][TILE_K];     //  8 KB
    __shared__ __bf16 lB[TILE_N][TILE_K];     // 16 KB
    __shared__ float  sw[TILE_M];

    const int tid   = threadIdx.x;
    const int mBase = blockIdx.x * TILE_M;
    const int dBase = blockIdx.y * TILE_N;
    if (mBase >= offs[E]) return;
    int e = 0;
    while (e < E - 1 && mBase >= offs[e + 1]) ++e;

    if (tid < TILE_M) sw[tid] = pw[mBase + tid];
    __syncthreads();

    const int wave = tid >> 5, lane = tid & 31;
    const int mSub = (wave >> 2) * 32, nSub = (wave & 3) * 32;
    v8f acc[2][2];
    #pragma unroll
    for (int mi = 0; mi < 2; ++mi)
        #pragma unroll
        for (int ni = 0; ni < 2; ++ni) acc[mi][ni] = (v8f){0,0,0,0,0,0,0,0};

    const int ar = tid >> 2, ac = tid & 3;
    const int br = tid >> 1, bc = tid & 1;
    const size_t wrow = ((size_t)e * D + dBase + br) * F;

    for (int k0 = 0; k0 < F; k0 += TILE_K) {
        // stage A: contiguous bf16 rows of h
        {
            const uint4* s = (const uint4*)(h + (size_t)(mBase + ar) * F + k0 + ac * 16);
            uint4* d = (uint4*)&lA[ar][ac * 16];
            d[0] = s[0]; d[1] = s[1];
        }
        // stage B: fp32 w2 -> bf16
        {
            const float* s = w2 + wrow + k0 + bc * 32;
            if (k0 + TILE_K < F) __builtin_prefetch(s + TILE_K, 0, 1);
            stage_row32(s, &lB[br][bc * 32]);
        }
        __syncthreads();
        #pragma unroll
        for (int ks = 0; ks < TILE_K; ks += 32) {
            const int akb = ks + ((lane & 16) ? 8 : 0);
            FragBF a[2];
            #pragma unroll
            for (int mi = 0; mi < 2; ++mi) {
                const int arow = mSub + mi * 16 + (lane & 15);
                a[mi].q[0] = *(const uint4*)&lA[arow][akb];
                a[mi].q[1] = *(const uint4*)&lA[arow][akb + 16];
            }
            const int bkb = ks + ((lane & 16) ? 8 : 0);
            FragBF b[2];
            #pragma unroll
            for (int ni = 0; ni < 2; ++ni) {
                const int bcol = nSub + ni * 16 + (lane & 15);
                b[ni].q[0] = *(const uint4*)&lB[bcol][bkb];
                b[ni].q[1] = *(const uint4*)&lB[bcol][bkb + 16];
            }
            #pragma unroll
            for (int mi = 0; mi < 2; ++mi)
                #pragma unroll
                for (int ni = 0; ni < 2; ++ni)
                    acc[mi][ni] = __builtin_amdgcn_wmma_f32_16x16x32_bf16(
                        false, a[mi].v, false, b[ni].v, (short)0, acc[mi][ni], false, false);
        }
        __syncthreads();
    }

    #pragma unroll
    for (int mi = 0; mi < 2; ++mi) {
        const int rowL = mSub + mi * 16 + ((lane & 16) ? 8 : 0);
        #pragma unroll
        for (int ni = 0; ni < 2; ++ni) {
            const int col = dBase + nSub + ni * 16 + (lane & 15);
            #pragma unroll
            for (int r = 0; r < 8; ++r)
                y[(size_t)(mBase + rowL + r) * D + col] = sw[rowL + r] * acc[mi][ni][r];
        }
    }
}

// -------------------- 7. combine the two routed contributions ---------------
__global__ __launch_bounds__(256) void combine_kernel(const float* __restrict__ y,
                                                      const int* __restrict__ ppos,
                                                      float* __restrict__ out) {
    int idx = blockIdx.x * 256 + threadIdx.x;            // float4 index over N*D
    int t = idx / (D / 4);
    int c = idx % (D / 4);
    int p0 = ppos[2*t], p1 = ppos[2*t + 1];
    const float4* y4 = (const float4*)y;
    float4 a = y4[(size_t)p0 * (D / 4) + c];
    float4 b = y4[(size_t)p1 * (D / 4) + c];
    float4 o; o.x = a.x + b.x; o.y = a.y + b.y; o.z = a.z + b.z; o.w = a.w + b.w;
    ((float4*)out)[idx] = o;
}

// ---------------------------------------------------------------------------
extern "C" void kernel_launch(void* const* d_in, const int* in_sizes, int n_in,
                              void* d_out, int out_size, void* d_ws, size_t ws_size,
                              hipStream_t stream) {
    (void)in_sizes; (void)n_in; (void)out_size; (void)ws_size;
    const float* stm    = (const float*)d_in[0];
    const float* gate_w = (const float*)d_in[1];
    const float* w1     = (const float*)d_in[2];
    const float* w2     = (const float*)d_in[3];
    const float* w3     = (const float*)d_in[4];
    float* out = (float*)d_out;

    char* p = (char*)d_ws;
    auto carve = [&](size_t bytes) -> char* {
        char* r = p; p += (bytes + 255) & ~(size_t)255; return r;
    };
    __bf16* xb     = (__bf16*)carve((size_t)N * D * 2);
    int*    topidx = (int*)   carve((size_t)N * TOPK * 4);
    float*  topw   = (float*) carve((size_t)N * TOPK * 4);
    int*    cnt    = (int*)   carve(E * 4);
    int*    cursor = (int*)   carve(E * 4);
    int*    offs   = (int*)   carve((E + 1) * 4);
    int*    ptok   = (int*)   carve((size_t)MAXPAIRS * 4);
    float*  pw     = (float*) carve((size_t)MAXPAIRS * 4);
    int*    ppos   = (int*)   carve((size_t)N * TOPK * 4);
    __bf16* hbuf   = (__bf16*)carve((size_t)MAXPAIRS * F * 2);
    float*  ybuf   = (float*) carve((size_t)MAXPAIRS * D * 4);

    prep_kernel   <<<(N * D / 4) / 256, 256, 0, stream>>>(stm, xb, cnt);
    router_kernel <<<N / 8,            256, 0, stream>>>(stm, gate_w, topidx, topw, cnt);
    offsets_kernel<<<1,                256, 0, stream>>>(cnt, offs, cursor, ptok, pw);
    scatter_kernel<<<N / 256,          256, 0, stream>>>(topidx, topw, cursor, ptok, pw, ppos);
    gemm1_kernel  <<<dim3(MTILES, F / TILE_N), 256, 0, stream>>>(xb, w1, w3, offs, ptok, hbuf);
    gemm2_kernel  <<<dim3(MTILES, D / TILE_N), 256, 0, stream>>>(hbuf, w2, offs, pw, ybuf);
    combine_kernel<<<(N * D / 4) / 256, 256, 0, stream>>>(ybuf, ppos, out);
}
